// HeteroGNN_74088185856217
// MI455X (gfx1250) — matrix operand order repaired and checked by reading, not compiled, not run on hardware
//
#include <hip/hip_runtime.h>
#include <hip/hip_bf16.h>

// ---------------------------------------------------------------------------
// HeteroGNN inference for MI455X (gfx1250).
//  - all GEMMs via v_wmma_f32_16x16x32_bf16 (bf16 operands, fp32 accumulate)
//  - mean aggregation via on-device CSR (built once, reused by both layers):
//    histogram -> single-block scan -> bucket fill -> gather-sum (no fp atomics)
//  - double-buffered WMMA inner loops (distinct regs per B subtile + k+32
//    prefetch) so wmmas don't stall on s_wait_loadcnt 0 per subtile
//   H=512, NP=20000, NS=10000, E=400000, FP=FS=64, NC=3
// ---------------------------------------------------------------------------

typedef __bf16 bf16;
typedef bf16  v16bf __attribute__((ext_vector_type(16)));
typedef bf16  v8bf  __attribute__((ext_vector_type(8)));
typedef float v8f   __attribute__((ext_vector_type(8)));

#define NPNODES 20000
#define NSNODES 10000
#define NEDGES  400000
#define HDIM    512
#define FDIM    64

__device__ __forceinline__ v8f wmma_bf16(v16bf a, v16bf b, v8f c) {
  // (neg_a, A, neg_b, B, c_mod, C, reuse_a, reuse_b)
  return __builtin_amdgcn_wmma_f32_16x16x32_bf16(false, a, false, b, (short)0, c,
                                                 false, false);
}

// A-operand fragment: elements 0..7 at p[0..7], elements 8..15 at p[16..23]
// (caller already applied the per-half-lane +8 element offset)
__device__ __forceinline__ v16bf load_a_frag(const bf16* p) {
  v8bf lo = *(const v8bf*)(p);
  v8bf hi = *(const v8bf*)(p + 16);
  return __builtin_shufflevector(lo, hi, 0,1,2,3,4,5,6,7,8,9,10,11,12,13,14,15);
}

// B-operand fragment: elements 0..7 at p[0..7], elements 8..15 at p[16..23]
// (caller applied +8 for upper half)
__device__ __forceinline__ v16bf load_b_frag(const bf16* p) {
  v8bf lo = *(const v8bf*)(p);
  v8bf hi = *(const v8bf*)(p + 16);
  return __builtin_shufflevector(lo, hi, 0,1,2,3,4,5,6,7,8,9,10,11,12,13,14,15);
}

// ----------------------------- small utilities -----------------------------

__global__ void k_cast_bf16(const float* __restrict__ in, bf16* __restrict__ out, int n) {
  int i = blockIdx.x * blockDim.x + threadIdx.x;
  if (i < n) out[i] = (bf16)in[i];
}

// in: fp32 row-major [K][N]  ->  out: bf16 row-major [N][K]  (W transposed)
__global__ void k_transpose_cast(const float* __restrict__ in, bf16* __restrict__ out,
                                 int K, int N) {
  int i = blockIdx.x * blockDim.x + threadIdx.x;
  if (i >= K * N) return;
  int n = i / K, k = i % K;
  out[(size_t)n * K + k] = (bf16)in[(size_t)k * N + n];
}

__global__ void k_zero_i32(int* __restrict__ p, int n) {
  int i = blockIdx.x * blockDim.x + threadIdx.x;
  if (i < n) p[i] = 0;
}

__global__ void k_copy_i32(const int* __restrict__ in, int* __restrict__ out, int n) {
  int i = blockIdx.x * blockDim.x + threadIdx.x;
  if (i < n) out[i] = in[i];
}

// ------------------------------- CSR build ---------------------------------

__global__ void k_hist(const int* __restrict__ dst, int* __restrict__ cnt, int n) {
  int e = blockIdx.x * blockDim.x + threadIdx.x;
  if (e < n) atomicAdd(&cnt[dst[e]], 1);
}

// single-block exclusive scan; start has n+1 entries (start[n] = total)
__global__ void k_scan_excl(const int* __restrict__ cnt, int* __restrict__ start, int n) {
  __shared__ int buf[1024];
  __shared__ int carry;
  if (threadIdx.x == 0) carry = 0;
  __syncthreads();
  for (int base = 0; base < n; base += 1024) {
    int i = base + (int)threadIdx.x;
    int v = (i < n) ? cnt[i] : 0;
    buf[threadIdx.x] = v;
    __syncthreads();
    for (int off = 1; off < 1024; off <<= 1) {
      int t = (threadIdx.x >= (unsigned)off) ? buf[threadIdx.x - off] : 0;
      __syncthreads();
      buf[threadIdx.x] += t;
      __syncthreads();
    }
    int incl = buf[threadIdx.x] + carry;
    if (i < n) start[i] = incl - v;  // exclusive
    __syncthreads();
    if (threadIdx.x == 1023) carry = incl;
    __syncthreads();
  }
  if (threadIdx.x == 0) start[n] = carry;
}

__global__ void k_fill(const int* __restrict__ src, const int* __restrict__ dst,
                       int* __restrict__ cursor, int* __restrict__ srcids, int n) {
  int e = blockIdx.x * blockDim.x + threadIdx.x;
  if (e < n) {
    int p = atomicAdd(&cursor[dst[e]], 1);
    srcids[p] = src[e];
  }
}

// gather-sum mean: one block (64 threads) per destination node; register acc,
// no feature atomics; writes mean directly in bf16.
__global__ void k_gather_mean(const bf16* __restrict__ X, const int* __restrict__ start,
                              const int* __restrict__ srcids, bf16* __restrict__ out) {
  int d  = blockIdx.x;
  int n0 = start[d], n1 = start[d + 1];
  v8f acc = {};
  for (int j = n0; j < n1; ++j) {
    const v8bf* row = (const v8bf*)(X + (size_t)srcids[j] * HDIM);
    v8bf v = row[threadIdx.x];
#pragma unroll
    for (int t = 0; t < 8; ++t) acc[t] += (float)v[t];
  }
  float inv = 1.0f / fmaxf((float)(n1 - n0), 1.0f);
  v8bf ov;
#pragma unroll
  for (int t = 0; t < 8; ++t) ov[t] = (bf16)(acc[t] * inv);
  *(v8bf*)(out + (size_t)d * HDIM + threadIdx.x * 8) = ov;
}

// --------------------------- fused WMMA GEMM -------------------------------
// C = relu( A1 @ W1^T + bias [+ A2 @ W2^T] )   bf16 operands, fp32 accumulate.
// The fused pair is treated as K-concatenation (virtual K' = 2K) so one
// double-buffered pipeline covers both. One wave = 16(M) x 64(N) tile.

__device__ __forceinline__ void gemm_fetch(v16bf& a, v16bf b[4],
                                           const bf16* a1, const bf16* a2,
                                           const bf16* W1, const bf16* W2,
                                           int K, int n0, int l15, int half, int kk) {
  const bf16* ab; const bf16* wb; int k;
  if (kk < K) { ab = a1; wb = W1; k = kk; }
  else        { ab = a2; wb = W2; k = kk - K; }
  a = load_a_frag(ab + k);
#pragma unroll
  for (int s = 0; s < 4; ++s)
    b[s] = load_b_frag(wb + (size_t)(n0 + s * 16 + l15) * K + k + half * 8);
}

__global__ void k_gemm_bias_relu(const bf16* __restrict__ A1, const bf16* __restrict__ W1,
                                 const bf16* __restrict__ A2, const bf16* __restrict__ W2,
                                 const float* __restrict__ bias, bf16* __restrict__ C,
                                 int M, int N, int K, int hasA2) {
  int wid  = (blockIdx.x * blockDim.x + threadIdx.x) >> 5;
  int lane = threadIdx.x & 31;
  int nT   = N >> 6;
  int m0   = (wid / nT) << 4;
  if (m0 >= M) return;
  int n0   = (wid % nT) << 6;
  int half = lane >> 4;
  int l15  = lane & 15;

  const bf16* a1 = A1 + (size_t)(m0 + l15) * K + half * 8;
  const bf16* a2 = (hasA2 ? A2 : A1) + (size_t)(m0 + l15) * K + half * 8;
  const int totalK = hasA2 ? 2 * K : K;  // always an even number of 32-steps here

  v8f acc[4] = {};
  v16bf a0f, a1f, b0f[4], b1f[4];
  gemm_fetch(a0f, b0f, a1, a2, W1, W2, K, n0, l15, half, 0);
  for (int kk = 0; kk < totalK; kk += 64) {
    gemm_fetch(a1f, b1f, a1, a2, W1, W2, K, n0, l15, half, kk + 32);
#pragma unroll
    for (int s = 0; s < 4; ++s) acc[s] = wmma_bf16(a0f, b0f[s], acc[s]);
    if (kk + 64 < totalK)
      gemm_fetch(a0f, b0f, a1, a2, W1, W2, K, n0, l15, half, kk + 64);
#pragma unroll
    for (int s = 0; s < 4; ++s) acc[s] = wmma_bf16(a1f, b1f[s], acc[s]);
  }

#pragma unroll
  for (int s = 0; s < 4; ++s) {
    int n = n0 + s * 16 + l15;
    float bv = bias[n];
#pragma unroll
    for (int i = 0; i < 8; ++i) {
      float v = acc[s][i] + bv;
      v = v > 0.0f ? v : 0.0f;
      C[(size_t)(m0 + i + half * 8) * N + n] = (bf16)v;
    }
  }
}

// ------------------------------ fused decoder ------------------------------
// out[e,0..2] = relu([hp2[row], hs2[col], attr] @ d1_W + d1_b) @ d2_W + d2_b
// K=1024 virtual (hp part = W1t cols 0..511, hs part = 512..1023); attr is a
// rank-2 fp32 update; second layer (N=3) folded in-register + shuffle reduce.

__device__ __forceinline__ void dec_fetch(v16bf& a, v16bf b[4],
                                          const bf16* ap, const bf16* as,
                                          const bf16* W1t, int n0, int l15, int half,
                                          int kk) {
  const bf16* arow = (kk < HDIM) ? (ap + kk) : (as + (kk - HDIM));
  a = load_a_frag(arow);
#pragma unroll
  for (int s = 0; s < 4; ++s)
    b[s] = load_b_frag(W1t + (size_t)(n0 + s * 16 + l15) * 1024 + kk + half * 8);
}

__global__ void k_decoder(const bf16* __restrict__ HP, const bf16* __restrict__ HS,
                          const bf16* __restrict__ W1t /*[512][1024]*/,
                          const float* __restrict__ d1W /*[1026][512]*/,
                          const float* __restrict__ d1b,
                          const float* __restrict__ d2W /*[512][3]*/,
                          const float* __restrict__ d2b,
                          const float* __restrict__ attr /*[E][2]*/,
                          const int* __restrict__ rows, const int* __restrict__ cols,
                          float* __restrict__ out, int E) {
  int wid  = (blockIdx.x * blockDim.x + threadIdx.x) >> 5;
  int lane = threadIdx.x & 31;
  int e0   = wid << 4;
  if (e0 >= E) return;
  int half = lane >> 4;
  int l15  = lane & 15;

  const bf16* ap = HP + (size_t)rows[e0 + l15] * HDIM + half * 8;
  const bf16* as = HS + (size_t)cols[e0 + l15] * HDIM + half * 8;

  int mbase = half * 8;
  float a0[8], a1v[8];
#pragma unroll
  for (int i = 0; i < 8; ++i) {
    int e = e0 + mbase + i;
    a0[i]  = attr[2 * e + 0];
    a1v[i] = attr[2 * e + 1];
  }

  float outacc[8][3] = {};

  for (int chunk = 0; chunk < 8; ++chunk) {
    int n0 = chunk * 64;
    v8f acc[4] = {};
    v16bf a0f, a1f, b0f[4], b1f[4];
    dec_fetch(a0f, b0f, ap, as, W1t, n0, l15, half, 0);
    for (int kk = 0; kk < 1024; kk += 64) {
      dec_fetch(a1f, b1f, ap, as, W1t, n0, l15, half, kk + 32);
#pragma unroll
      for (int s = 0; s < 4; ++s) acc[s] = wmma_bf16(a0f, b0f[s], acc[s]);
      if (kk + 64 < 1024)
        dec_fetch(a0f, b0f, ap, as, W1t, n0, l15, half, kk + 64);
#pragma unroll
      for (int s = 0; s < 4; ++s) acc[s] = wmma_bf16(a1f, b1f[s], acc[s]);
    }
#pragma unroll
    for (int s = 0; s < 4; ++s) {
      int n = n0 + s * 16 + l15;
      float bn = d1b[n];
      float w0 = d1W[(size_t)1024 * HDIM + n];  // attr weight row 0
      float w1 = d1W[(size_t)1025 * HDIM + n];  // attr weight row 1
      float wc0 = d2W[n * 3 + 0], wc1 = d2W[n * 3 + 1], wc2 = d2W[n * 3 + 2];
#pragma unroll
      for (int i = 0; i < 8; ++i) {
        float h = acc[s][i] + bn + a0[i] * w0 + a1v[i] * w1;
        h = h > 0.0f ? h : 0.0f;
        outacc[i][0] += h * wc0;
        outacc[i][1] += h * wc1;
        outacc[i][2] += h * wc2;
      }
    }
  }

  // reduce over the 16 N-lanes of each half (xor 1,2,4,8 stays within a half)
#pragma unroll
  for (int i = 0; i < 8; ++i) {
#pragma unroll
    for (int c = 0; c < 3; ++c) {
      float v = outacc[i][c];
      v += __shfl_xor(v, 1);
      v += __shfl_xor(v, 2);
      v += __shfl_xor(v, 4);
      v += __shfl_xor(v, 8);
      outacc[i][c] = v;
    }
  }
  if (l15 == 0) {
#pragma unroll
    for (int i = 0; i < 8; ++i)
#pragma unroll
      for (int c = 0; c < 3; ++c)
        out[(size_t)(e0 + mbase + i) * 3 + c] = outacc[i][c] + d2b[c];
  }
}

// ------------------------------ host driver --------------------------------

extern "C" void kernel_launch(void* const* d_in, const int* in_sizes, int n_in,
                              void* d_out, int out_size, void* d_ws, size_t ws_size,
                              hipStream_t stream) {
  const float* x_pol  = (const float*)d_in[0];
  const float* x_stk  = (const float*)d_in[1];
  const float* attr   = (const float*)d_in[2];
  const float* Wp     = (const float*)d_in[3];
  const float* bp     = (const float*)d_in[4];
  const float* Ws     = (const float*)d_in[5];
  const float* bs     = (const float*)d_in[6];
  const float* cW[12];
  for (int i = 0; i < 12; ++i) cW[i] = (const float*)d_in[7 + i];
  // cW: [c1_ps_Wl, c1_ps_bl, c1_ps_Wr, c1_sp_Wl, c1_sp_bl, c1_sp_Wr,
  //      c2_ps_Wl, c2_ps_bl, c2_ps_Wr, c2_sp_Wl, c2_sp_bl, c2_sp_Wr]
  const float* d1_W   = (const float*)d_in[19];
  const float* d1_b   = (const float*)d_in[20];
  const float* d2_W   = (const float*)d_in[21];
  const float* d2_b   = (const float*)d_in[22];
  const int* eps_src  = (const int*)d_in[23];
  const int* eps_dst  = (const int*)d_in[24];
  const int* esp_src  = (const int*)d_in[25];
  const int* esp_dst  = (const int*)d_in[26];
  const int* trow     = (const int*)d_in[27];
  const int* tcol     = (const int*)d_in[28];
  float* out          = (float*)d_out;
  (void)in_sizes; (void)n_in; (void)out_size; (void)ws_size;

  // ---- workspace carve-out ----
  size_t off = 0;
  auto take = [&](size_t bytes) -> void* {
    off = (off + 255) & ~(size_t)255;
    void* p = (char*)d_ws + off;
    off += bytes;
    return p;
  };
  bf16* xp_b    = (bf16*)take((size_t)NPNODES * FDIM * 2);
  bf16* xs_b    = (bf16*)take((size_t)NSNODES * FDIM * 2);
  bf16* WpT     = (bf16*)take((size_t)HDIM * FDIM * 2);
  bf16* WsT     = (bf16*)take((size_t)HDIM * FDIM * 2);
  bf16* convWT[8];  // ps_Wl, ps_Wr, sp_Wl, sp_Wr for c1 then c2
  for (int i = 0; i < 8; ++i) convWT[i] = (bf16*)take((size_t)HDIM * HDIM * 2);
  bf16* d1Wt    = (bf16*)take((size_t)HDIM * 1024 * 2);
  bf16* hp_b    = (bf16*)take((size_t)NPNODES * HDIM * 2);  // hp, later hp2
  bf16* hs_b    = (bf16*)take((size_t)NSNODES * HDIM * 2);  // hs, later hs2
  bf16* hp1_b   = (bf16*)take((size_t)NPNODES * HDIM * 2);
  bf16* hs1_b   = (bf16*)take((size_t)NSNODES * HDIM * 2);
  bf16* mean_pb = (bf16*)take((size_t)NPNODES * HDIM * 2);
  bf16* mean_sb = (bf16*)take((size_t)NSNODES * HDIM * 2);
  int* cnt_i    = (int*)take((size_t)NPNODES * 4);
  int* start_s  = (int*)take((size_t)(NSNODES + 1) * 4);
  int* start_p  = (int*)take((size_t)(NPNODES + 1) * 4);
  int* cursor   = (int*)take((size_t)NPNODES * 4);
  int* srcs_ps  = (int*)take((size_t)NEDGES * 4);  // CSR by stock dst (pol src ids)
  int* srcs_sp  = (int*)take((size_t)NEDGES * 4);  // CSR by pol dst (stock src ids)

  auto cdiv = [](int a, int b) { return (a + b - 1) / b; };

  // ---- 1) cast inputs, transpose+cast weights ----
  k_cast_bf16<<<cdiv(NPNODES * FDIM, 256), 256, 0, stream>>>(x_pol, xp_b, NPNODES * FDIM);
  k_cast_bf16<<<cdiv(NSNODES * FDIM, 256), 256, 0, stream>>>(x_stk, xs_b, NSNODES * FDIM);
  k_transpose_cast<<<cdiv(FDIM * HDIM, 256), 256, 0, stream>>>(Wp, WpT, FDIM, HDIM);
  k_transpose_cast<<<cdiv(FDIM * HDIM, 256), 256, 0, stream>>>(Ws, WsT, FDIM, HDIM);
  const float* convSrc[8] = {cW[0], cW[2], cW[3], cW[5], cW[6], cW[8], cW[9], cW[11]};
  for (int i = 0; i < 8; ++i)
    k_transpose_cast<<<cdiv(HDIM * HDIM, 256), 256, 0, stream>>>(convSrc[i], convWT[i],
                                                                 HDIM, HDIM);
  k_transpose_cast<<<cdiv(1024 * HDIM, 256), 256, 0, stream>>>(d1_W, d1Wt, 1024, HDIM);

  // ---- 2) build both CSRs once (reused by both layers) ----
  auto build_csr = [&](const int* src, const int* dst, int nDst, int* start, int* srcids) {
    k_zero_i32<<<cdiv(nDst, 256), 256, 0, stream>>>(cnt_i, nDst);
    k_hist<<<cdiv(NEDGES, 256), 256, 0, stream>>>(dst, cnt_i, NEDGES);
    k_scan_excl<<<1, 1024, 0, stream>>>(cnt_i, start, nDst);
    k_copy_i32<<<cdiv(nDst, 256), 256, 0, stream>>>(start, cursor, nDst);
    k_fill<<<cdiv(NEDGES, 256), 256, 0, stream>>>(src, dst, cursor, srcids, NEDGES);
  };
  build_csr(eps_src, eps_dst, NSNODES, start_s, srcs_ps);
  build_csr(esp_src, esp_dst, NPNODES, start_p, srcs_sp);

  auto gemm = [&](const bf16* A1, const bf16* W1, const bf16* A2, const bf16* W2,
                  const float* bias, bf16* C, int M, int N, int K, int hasA2) {
    int waves = (M / 16) * (N / 64);
    k_gemm_bias_relu<<<cdiv(waves, 8), 256, 0, stream>>>(A1, W1, A2, W2, bias, C,
                                                         M, N, K, hasA2);
  };

  // ---- 3) input projections ----
  gemm(xp_b, WpT, nullptr, nullptr, bp, hp_b, NPNODES, HDIM, FDIM, 0);
  gemm(xs_b, WsT, nullptr, nullptr, bs, hs_b, NSNODES, HDIM, FDIM, 0);

  // ---- 4) two SAGE layers ----
  auto layer = [&](const bf16* hp_in, const bf16* hs_in,
                   const bf16* psWlT, const float* ps_bl, const bf16* psWrT,
                   const bf16* spWlT, const float* sp_bl, const bf16* spWrT,
                   bf16* hp_out, bf16* hs_out) {
    k_gather_mean<<<NSNODES, 64, 0, stream>>>(hp_in, start_s, srcs_ps, mean_sb);
    k_gather_mean<<<NPNODES, 64, 0, stream>>>(hs_in, start_p, srcs_sp, mean_pb);
    // hs_out = relu(mean_s @ ps_Wl + ps_bl + hs_in @ ps_Wr)
    gemm(mean_sb, psWlT, hs_in, psWrT, ps_bl, hs_out, NSNODES, HDIM, HDIM, 1);
    // hp_out = relu(mean_p @ sp_Wl + sp_bl + hp_in @ sp_Wr)
    gemm(mean_pb, spWlT, hp_in, spWrT, sp_bl, hp_out, NPNODES, HDIM, HDIM, 1);
  };

  layer(hp_b, hs_b, convWT[0], cW[1], convWT[1], convWT[2], cW[4], convWT[3],
        hp1_b, hs1_b);
  // layer 2 writes hp2/hs2 into hp_b/hs_b (safe: hp_b/hs_b dead after layer 1)
  layer(hp1_b, hs1_b, convWT[4], cW[7], convWT[5], convWT[6], cW[10], convWT[7],
        hp_b, hs_b);

  // ---- 5) fused edge decoder ----
  int dwaves = NEDGES / 16;
  k_decoder<<<cdiv(dwaves, 8), 256, 0, stream>>>(hp_b, hs_b, d1Wt, d1_W, d1_b,
                                                 d2_W, d2_b, attr, trow, tcol,
                                                 out, NEDGES);
}
